// DeepSeekLayer_4879082848969
// MI455X (gfx1250) — compile-verified, matrix operand order checked by reading, not run on hardware
//
#include <hip/hip_runtime.h>

// ---------------------------------------------------------------------------
// DeepSeek layer (MLA attention + MoE) for gfx1250 (MI455X, wave32, WMMA).
// GEMMs: 128x128 block tile, 8 waves x (32x64) => 8 v_wmma per wave per K-step,
// double-buffered LDS staged with global_load_async_to_lds_b128 (ASYNCcnt).
// ---------------------------------------------------------------------------

typedef _Float16 half16_t __attribute__((ext_vector_type(16)));
typedef float    float8_t __attribute__((ext_vector_type(8)));
typedef __attribute__((address_space(3))) _Float16 lds_f16;

#define TM 128
#define TN 128
#define TK 32

// async DMA: 16 bytes global -> LDS, tracked by ASYNCcnt
__device__ __forceinline__ void async_b128(const _Float16* g, _Float16* l)
{
    unsigned lofs = (unsigned)(unsigned long long)(lds_f16*)l;  // LDS byte addr
    asm volatile("global_load_async_to_lds_b128 %0, %1, off"
                 :: "v"(lofs), "v"(g)
                 : "memory");
}
__device__ __forceinline__ void wait_async0()
{
    asm volatile("s_wait_asynccnt 0x0" ::: "memory");
}

// ---------------------------------------------------------------------------
// Generic f16 WMMA GEMM:  C[M,N] = alpha * A[M,K] . B^T  (+ C if beta)
//   b_kn == 0 : B stored row-major [N,K]  (torch Linear weight, y = x @ W.T)
//   b_kn == 1 : B stored row-major [K,N]  (plain matmul, e.g. attn @ V)
// M % 128 == 0, N % 128 == 0, K % 32 == 0 (true for all shapes here).
// ---------------------------------------------------------------------------
__global__ __launch_bounds__(256)
void k_gemm_f16(const _Float16* __restrict__ A, const _Float16* __restrict__ B,
                float* __restrict__ C,
                int M, int N, int K, int lda, int ldb, int ldc,
                float alpha, int beta, int b_kn)
{
    __shared__ __align__(64) _Float16 As[2][TM][TK];   // 16 KB
    __shared__ __align__(64) _Float16 Bs[2][TN][TK];   // 16 KB

    const int tid  = threadIdx.x;
    const int wave = tid >> 5;
    const int lane = tid & 31;
    const int m0   = blockIdx.y * TM;
    const int n0   = blockIdx.x * TN;
    const int wm   = (wave & 3) * 32;    // wave's 32-row strip
    const int wn   = (wave >> 2) * 64;   // wave's 64-col strip

    float8_t acc[2][4] = {};

    // staging maps: 128 rows x 32 cols f16 per tile, 256 threads x 32B each
    const int lr = tid >> 1;             // 0..127 row
    const int lc = (tid & 1) * 16;       // 0 or 16 (two b128 per thread)
    // b_kn transpose-scatter map
    const int kk = tid & 31;             // 0..31 K row
    const int nb = (tid >> 5) * 16;      // 0..112 N base (16 per thread)

    auto stageA = [&](int k0, int buf) {
        const _Float16* g = &A[(long)(m0 + lr) * lda + k0 + lc];
        async_b128(g,     &As[buf][lr][lc]);
        async_b128(g + 8, &As[buf][lr][lc + 8]);
    };
    auto stageB = [&](int k0, int buf) {
        if (!b_kn) {
            const _Float16* g = &B[(long)(n0 + lr) * ldb + k0 + lc];
            async_b128(g,     &Bs[buf][lr][lc]);
            async_b128(g + 8, &Bs[buf][lr][lc + 8]);
        } else {
            const _Float16* g = &B[(long)(k0 + kk) * ldb + n0 + nb];
            __builtin_prefetch(g + ldb * TK, 0, 0);
            #pragma unroll
            for (int i = 0; i < 16; ++i) Bs[buf][nb + i][kk] = g[i];
        }
    };

    // prologue: stage tile 0
    stageA(0, 0);
    stageB(0, 0);
    wait_async0();
    __syncthreads();

    const int l16 = lane & 15;
    const int kh  = (lane >> 4) * 16;

    for (int k0 = 0; k0 < K; k0 += TK) {
        const int buf = (k0 / TK) & 1;
        const int k1  = k0 + TK;
        if (k1 < K) {                    // overlap next tile's DMA with compute
            stageA(k1, buf ^ 1);
            stageB(k1, buf ^ 1);
        }

        // fragments (16 f16 per lane, 32B LDS reads)
        const _Float16* pa = &As[buf][wm + l16][kh];
        half16_t a0 = *(const half16_t*)pa;
        half16_t a1 = *(const half16_t*)(pa + 16 * TK);
        const _Float16* pb = &Bs[buf][wn + l16][kh];
        half16_t b0 = *(const half16_t*)pb;
        half16_t b1 = *(const half16_t*)(pb + 16 * TK);
        half16_t b2 = *(const half16_t*)(pb + 32 * TK);
        half16_t b3 = *(const half16_t*)(pb + 48 * TK);

        acc[0][0] = __builtin_amdgcn_wmma_f32_16x16x32_f16(false, a0, false, b0, (short)0, acc[0][0], false, false);
        acc[0][1] = __builtin_amdgcn_wmma_f32_16x16x32_f16(false, a0, false, b1, (short)0, acc[0][1], false, false);
        acc[0][2] = __builtin_amdgcn_wmma_f32_16x16x32_f16(false, a0, false, b2, (short)0, acc[0][2], false, false);
        acc[0][3] = __builtin_amdgcn_wmma_f32_16x16x32_f16(false, a0, false, b3, (short)0, acc[0][3], false, false);
        acc[1][0] = __builtin_amdgcn_wmma_f32_16x16x32_f16(false, a1, false, b0, (short)0, acc[1][0], false, false);
        acc[1][1] = __builtin_amdgcn_wmma_f32_16x16x32_f16(false, a1, false, b1, (short)0, acc[1][1], false, false);
        acc[1][2] = __builtin_amdgcn_wmma_f32_16x16x32_f16(false, a1, false, b2, (short)0, acc[1][2], false, false);
        acc[1][3] = __builtin_amdgcn_wmma_f32_16x16x32_f16(false, a1, false, b3, (short)0, acc[1][3], false, false);

        wait_async0();                   // DMA for tile k+1 done
        __syncthreads();                 // all waves done reading tile k
    }

    // epilogue: D layout -> VGPR g: lanes 0-15 M=g, lanes 16-31 M=8+g
    const int rsel = (lane >> 4) ? 8 : 0;
    #pragma unroll
    for (int i = 0; i < 2; ++i) {
        #pragma unroll
        for (int g = 0; g < 8; ++g) {
            const long row = m0 + wm + i * 16 + rsel + g;
            float* cr = C + row * (long)ldc + n0 + wn + l16;
            #pragma unroll
            for (int j = 0; j < 4; ++j) {
                float v = alpha * acc[i][j][g];
                if (beta) v += cr[j * 16];
                cr[j * 16] = v;
            }
        }
    }
}

// ---------------------------------------------------------------------------
// fp32 -> f16 convert
// ---------------------------------------------------------------------------
__global__ void k_f32_to_f16(const float* __restrict__ src,
                             _Float16* __restrict__ dst, long n)
{
    long i = (long)blockIdx.x * blockDim.x + threadIdx.x;
    if (i < n) dst[i] = (_Float16)src[i];
}

// ---------------------------------------------------------------------------
// RMSNorm row kernel: out_f16[row] = x * rsqrt(mean(x^2)+eps) * g
// ---------------------------------------------------------------------------
__global__ __launch_bounds__(256)
void k_rmsnorm_f16(const float* __restrict__ x, const float* __restrict__ g,
                   _Float16* __restrict__ out, int d)
{
    __shared__ float red[8];
    const long row = blockIdx.x;
    const float* xr = x + row * (long)d;

    float ss = 0.f;
    for (int i = threadIdx.x; i < d; i += 256) { float v = xr[i]; ss += v * v; }
    #pragma unroll
    for (int o = 16; o > 0; o >>= 1) ss += __shfl_down(ss, o, 32);
    if ((threadIdx.x & 31) == 0) red[threadIdx.x >> 5] = ss;
    __syncthreads();
    if (threadIdx.x == 0) {
        float t = 0.f;
        #pragma unroll
        for (int i = 0; i < 8; ++i) t += red[i];
        red[0] = t;
    }
    __syncthreads();
    const float inv = rsqrtf(red[0] / d + 1e-6f);
    for (int i = threadIdx.x; i < d; i += 256)
        out[row * (long)d + i] = (_Float16)(xr[i] * inv * g[i]);
}

// ---------------------------------------------------------------------------
// row softmax (f32 in -> f16 out)
// ---------------------------------------------------------------------------
__global__ __launch_bounds__(256)
void k_softmax_f16(const float* __restrict__ s, _Float16* __restrict__ out, int n)
{
    __shared__ float red[8];
    __shared__ float stat[2];
    const long row = blockIdx.x;
    const float* sr = s + row * (long)n;

    float mx = -3.4e38f;
    for (int i = threadIdx.x; i < n; i += 256) mx = fmaxf(mx, sr[i]);
    #pragma unroll
    for (int o = 16; o > 0; o >>= 1) mx = fmaxf(mx, __shfl_down(mx, o, 32));
    if ((threadIdx.x & 31) == 0) red[threadIdx.x >> 5] = mx;
    __syncthreads();
    if (threadIdx.x == 0) {
        float m = red[0];
        #pragma unroll
        for (int i = 1; i < 8; ++i) m = fmaxf(m, red[i]);
        stat[0] = m;
    }
    __syncthreads();
    const float m = stat[0];

    float sum = 0.f;
    for (int i = threadIdx.x; i < n; i += 256) sum += __expf(sr[i] - m);
    #pragma unroll
    for (int o = 16; o > 0; o >>= 1) sum += __shfl_down(sum, o, 32);
    if ((threadIdx.x & 31) == 0) red[threadIdx.x >> 5] = sum;
    __syncthreads();
    if (threadIdx.x == 0) {
        float t = 0.f;
        #pragma unroll
        for (int i = 0; i < 8; ++i) t += red[i];
        stat[1] = t;
    }
    __syncthreads();
    const float inv = 1.f / stat[1];
    for (int i = threadIdx.x; i < n; i += 256)
        out[row * (long)n + i] = (_Float16)(__expf(sr[i] - m) * inv);
}

// ---------------------------------------------------------------------------
// MoE router: logits = nx @ Wr.T + bias; softmax; top-2; softmax over top-2
// ---------------------------------------------------------------------------
__global__ __launch_bounds__(256)
void k_router(const _Float16* __restrict__ nx, const float* __restrict__ Wr,
              const float* __restrict__ bias, float* __restrict__ combine, int d)
{
    __shared__ float lg[8];
    const long t   = blockIdx.x;
    const int wave = threadIdx.x >> 5;
    const int lane = threadIdx.x & 31;

    const _Float16* xr = nx + t * (long)d;
    const float*    wr = Wr + (long)wave * d;
    float acc = 0.f;
    for (int i = lane; i < d; i += 32) acc += (float)xr[i] * wr[i];
    #pragma unroll
    for (int o = 16; o > 0; o >>= 1) acc += __shfl_down(acc, o, 32);
    if (lane == 0) lg[wave] = acc + bias[wave];
    __syncthreads();

    if (threadIdx.x == 0) {
        float m = lg[0];
        #pragma unroll
        for (int i = 1; i < 8; ++i) m = fmaxf(m, lg[i]);
        float e[8], s = 0.f;
        #pragma unroll
        for (int i = 0; i < 8; ++i) { e[i] = __expf(lg[i] - m); s += e[i]; }
        #pragma unroll
        for (int i = 0; i < 8; ++i) e[i] /= s;
        int i0 = 0;
        for (int i = 1; i < 8; ++i) if (e[i] > e[i0]) i0 = i;
        int i1 = (i0 == 0) ? 1 : 0;
        for (int i = 0; i < 8; ++i) { if (i == i0) continue; if (e[i] > e[i1]) i1 = i; }
        const float a = __expf(e[i0]), b = __expf(e[i1]);  // softmax over top-2
        #pragma unroll
        for (int i = 0; i < 8; ++i) combine[t * 8 + i] = 0.f;
        combine[t * 8 + i0] = a / (a + b);
        combine[t * 8 + i1] = b / (a + b);
    }
}

// ---------------------------------------------------------------------------
// h = silu(t1) * t3 [* rowscale[row]]  -> f16
// ---------------------------------------------------------------------------
__global__ void k_silu_mul(const float* __restrict__ t1, const float* __restrict__ t3,
                           const float* __restrict__ rs, int rs_stride,
                           _Float16* __restrict__ out, long n, int rowlen)
{
    long i = (long)blockIdx.x * blockDim.x + threadIdx.x;
    if (i >= n) return;
    const float a = t1[i];
    float v = (a / (1.f + __expf(-a))) * t3[i];
    if (rs) v *= rs[(i / rowlen) * (long)rs_stride];
    out[i] = (_Float16)v;
}

// ---------------------------------------------------------------------------
// host-side launch
// ---------------------------------------------------------------------------
extern "C" void kernel_launch(void* const* d_in, const int* in_sizes, int n_in,
                              void* d_out, int out_size, void* d_ws, size_t ws_size,
                              hipStream_t stream)
{
    (void)in_sizes; (void)n_in; (void)out_size; (void)ws_size;

    const float* X    = (const float*)d_in[0];
    const float* g1   = (const float*)d_in[1];
    const float* g2   = (const float*)d_in[2];
    const float* Wq   = (const float*)d_in[3];
    const float* Wkc  = (const float*)d_in[4];
    const float* Wvc  = (const float*)d_in[5];
    const float* Wo   = (const float*)d_in[6];
    const float* Wr   = (const float*)d_in[7];
    const float* ebia = (const float*)d_in[8];
    const float* sW1  = (const float*)d_in[9];
    const float* sW2  = (const float*)d_in[10];
    const float* sW3  = (const float*)d_in[11];
    const float* eW1  = (const float*)d_in[12];
    const float* eW2  = (const float*)d_in[13];
    const float* eW3  = (const float*)d_in[14];
    float* out = (float*)d_out;

    constexpr int  Bb = 2, S = 1024, d = 1024, H = 4, dk = 256, dc = 256, E = 8, f = 1024;
    constexpr long T  = (long)Bb * S;
    const float scale = 1.0f / 16.0f;   // 1/sqrt(dk)

    // ---- workspace carving (256B aligned bump allocator) ----
    char* ws = (char*)d_ws;
    size_t off = 0;
    auto alloc = [&](size_t bytes) -> char* {
        char* p = ws + off;
        off += (bytes + 255) & ~(size_t)255;
        return p;
    };
    // f16 weight copies
    _Float16* wqh  = (_Float16*)alloc((size_t)d * d * 2);
    _Float16* wkch = (_Float16*)alloc((size_t)dc * d * 2);
    _Float16* wvch = (_Float16*)alloc((size_t)dc * d * 2);
    _Float16* woh  = (_Float16*)alloc((size_t)d * d * 2);
    _Float16* sw1h = (_Float16*)alloc((size_t)f * d * 2);
    _Float16* sw2h = (_Float16*)alloc((size_t)d * f * 2);
    _Float16* sw3h = (_Float16*)alloc((size_t)f * d * 2);
    _Float16* ew1h = (_Float16*)alloc((size_t)E * f * d * 2);
    _Float16* ew2h = (_Float16*)alloc((size_t)E * d * f * 2);
    _Float16* ew3h = (_Float16*)alloc((size_t)E * f * d * 2);
    // attention-phase activations
    _Float16* nx1h   = (_Float16*)alloc((size_t)T * d * 2);
    float*    qf     = (float*)   alloc((size_t)T * d * 4);
    _Float16* qh     = (_Float16*)alloc((size_t)T * d * 2);
    float*    kcf    = (float*)   alloc((size_t)T * dc * 4);
    _Float16* kch    = (_Float16*)alloc((size_t)T * dc * 2);
    float*    vcf    = (float*)   alloc((size_t)T * dc * 4);
    _Float16* vch    = (_Float16*)alloc((size_t)T * dc * 2);
    float*    scores = (float*)   alloc((size_t)Bb * H * S * S * 4);   // 32 MB
    _Float16* attnh  = (_Float16*)alloc((size_t)Bb * H * S * S * 2);   // 16 MB
    float*    aof    = (float*)   alloc((size_t)T * d * 4);
    _Float16* aoh    = (_Float16*)alloc((size_t)T * d * 2);
    // MoE-phase buffers alias the (dead-by-then) scores+attnh region (48 MB)
    {
        char* moe = (char*)scores;
        size_t mo = 0;
        auto malloc2 = [&](size_t bytes) -> char* {
            char* p = moe + mo;
            mo += (bytes + 255) & ~(size_t)255;
            return p;
        };
        float*    xr_      = (float*)   malloc2((size_t)T * d * 4);
        _Float16* nx2h_    = (_Float16*)malloc2((size_t)T * d * 2);
        float*    t1_      = (float*)   malloc2((size_t)T * f * 4);
        float*    t3_      = (float*)   malloc2((size_t)T * f * 4);
        _Float16* hh_      = (_Float16*)malloc2((size_t)T * f * 2);
        float*    combine_ = (float*)   malloc2((size_t)T * E * 4);

        // ---- helpers ----
        auto conv = [&](const float* s, _Float16* dst, long n) {
            k_f32_to_f16<<<(int)((n + 255) / 256), 256, 0, stream>>>(s, dst, n);
        };
        auto gemm = [&](const _Float16* A, const _Float16* Bm, float* C,
                        int M, int N, int K, int lda, int ldb, int ldc,
                        float alpha, int beta, int bkn) {
            dim3 g(N / TN, M / TM);
            k_gemm_f16<<<g, 256, 0, stream>>>(A, Bm, C, M, N, K, lda, ldb, ldc,
                                              alpha, beta, bkn);
        };

        // ---- 0. weights fp32 -> f16 ----
        conv(Wq,  wqh,  (long)d * d);
        conv(Wkc, wkch, (long)dc * d);
        conv(Wvc, wvch, (long)dc * d);
        conv(Wo,  woh,  (long)d * d);
        conv(sW1, sw1h, (long)f * d);
        conv(sW2, sw2h, (long)d * f);
        conv(sW3, sw3h, (long)f * d);
        conv(eW1, ew1h, (long)E * f * d);
        conv(eW2, ew2h, (long)E * d * f);
        conv(eW3, ew3h, (long)E * f * d);

        // ---- 1. MLA attention ----
        k_rmsnorm_f16<<<(int)T, 256, 0, stream>>>(X, g1, nx1h, d);

        gemm(nx1h, wqh,  qf,  (int)T, d,  d, d, d, d,  1.f, 0, 0);   // Q
        gemm(nx1h, wkch, kcf, (int)T, dc, d, d, d, dc, 1.f, 0, 0);   // Kc
        gemm(nx1h, wvch, vcf, (int)T, dc, d, d, d, dc, 1.f, 0, 0);   // Vc
        conv(qf,  qh,  (long)T * d);
        conv(kcf, kch, (long)T * dc);
        conv(vcf, vch, (long)T * dc);

        for (int b = 0; b < Bb; ++b)
            for (int h = 0; h < H; ++h) {
                const _Float16* Aq = qh  + (long)b * S * d + (long)h * dk;
                const _Float16* Bk = kch + (long)b * S * dc;
                float* Cs = scores + ((long)b * H + h) * S * S;
                gemm(Aq, Bk, Cs, S, S, dk, d, dc, S, scale, 0, 0);   // scores
            }

        k_softmax_f16<<<Bb * H * S, 256, 0, stream>>>(scores, attnh, S);

        for (int b = 0; b < Bb; ++b)
            for (int h = 0; h < H; ++h) {
                const _Float16* Aa = attnh + ((long)b * H + h) * S * S;
                const _Float16* Bv = vch + (long)b * S * dc;
                float* Co = aof + (long)b * S * d + (long)h * dk;
                gemm(Aa, Bv, Co, S, dc, S, S, dc, d, 1.f, 0, 1);     // attn @ V
            }
        conv(aof, aoh, (long)T * d);

        // xr = X + ao @ Wo.T
        hipMemcpyAsync(xr_, X, (size_t)T * d * 4, hipMemcpyDeviceToDevice, stream);
        gemm(aoh, woh, xr_, (int)T, d, d, d, d, d, 1.f, 1, 0);

        // ---- 2. MoE ----
        k_rmsnorm_f16<<<(int)T, 256, 0, stream>>>(xr_, g2, nx2h_, d);
        k_router<<<(int)T, 256, 0, stream>>>(nx2h_, Wr, ebia, combine_, d);

        // out = xr (residual), then accumulate shared + experts with beta=1
        hipMemcpyAsync(out, xr_, (size_t)T * d * 4, hipMemcpyDeviceToDevice, stream);

        const long nTf = (long)T * f;
        // shared expert
        gemm(nx2h_, sw1h, t1_, (int)T, f, d, d, d, f, 1.f, 0, 0);
        gemm(nx2h_, sw3h, t3_, (int)T, f, d, d, d, f, 1.f, 0, 0);
        k_silu_mul<<<(int)((nTf + 255) / 256), 256, 0, stream>>>(
            t1_, t3_, (const float*)nullptr, 0, hh_, nTf, f);
        gemm(hh_, sw2h, out, (int)T, d, f, f, f, d, 1.f, 1, 0);

        // routed experts (dense over all tokens; combine folded into h)
        for (int e = 0; e < E; ++e) {
            const _Float16* w1 = ew1h + (long)e * f * d;
            const _Float16* w3 = ew3h + (long)e * f * d;
            const _Float16* w2 = ew2h + (long)e * d * f;
            gemm(nx2h_, w1, t1_, (int)T, f, d, d, d, f, 1.f, 0, 0);
            gemm(nx2h_, w3, t3_, (int)T, f, d, d, d, f, 1.f, 0, 0);
            k_silu_mul<<<(int)((nTf + 255) / 256), 256, 0, stream>>>(
                t1_, t3_, combine_ + e, E, hh_, nTf, f);
            gemm(hh_, w2, out, (int)T, d, f, f, f, d, 1.f, 1, 0);
        }
    }
}
